// CoreAttentionExpand_42880953484143
// MI455X (gfx1250) — compile-verified, matrix operand order checked by reading, not run on hardware
//
#include <hip/hip_runtime.h>
#include <math.h>

typedef _Float16 v16h __attribute__((ext_vector_type(16)));
typedef _Float16 v8h  __attribute__((ext_vector_type(8)));
typedef float    v8f  __attribute__((ext_vector_type(8)));

#define NHEAD 16
#define TLEN  4096
#define DDIM  128
#define LCHUNK 1024
#define EHIS  128
#define PNUM  4
#define MLEN  1152   /* EHIS + LCHUNK */

__device__ __forceinline__ v8h ld8(const _Float16* p){ return *(const v8h*)p; }
__device__ __forceinline__ v16h cat8(v8h lo, v8h hi){
  return __builtin_shufflevector(lo, hi, 0,1,2,3,4,5,6,7,8,9,10,11,12,13,14,15);
}
__device__ __forceinline__ v8f wmma16(v16h a, v16h b, v8f c){
  return __builtin_amdgcn_wmma_f32_16x16x32_f16(false, a, false, b, (short)0, c, false, false);
}

// DPP16 cross-lane (stays in VALU pipe; co-executes with WMMA, no DS round-trip).
// Patterns cover a full 16-lane butterfly: quad xor1 (0xB1), quad xor2 (0x4E),
// row_half_mirror (0x141), row_mirror (0x140).
template<int CTRL>
__device__ __forceinline__ float dppf(float x){
  return __builtin_bit_cast(float,
      __builtin_amdgcn_mov_dpp(__builtin_bit_cast(int, x), CTRL, 0xf, 0xf, true));
}
__device__ __forceinline__ float rmax16(float x){
  x = fmaxf(x, dppf<0xB1>(x));
  x = fmaxf(x, dppf<0x4E>(x));
  x = fmaxf(x, dppf<0x141>(x));
  x = fmaxf(x, dppf<0x140>(x));
  return x;
}
__device__ __forceinline__ float rsum16(float x){
  x += dppf<0xB1>(x);
  x += dppf<0x4E>(x);
  x += dppf<0x141>(x);
  x += dppf<0x140>(x);
  return x;
}

// ---------------- prep: RoPE + f16 staging (QH, expanded KL, transposed VT) ----------------
__global__ __launch_bounds__(256) void prep_kernel(
    const float* __restrict__ q, const float* __restrict__ k, const float* __restrict__ v,
    _Float16* __restrict__ QH, _Float16* __restrict__ KL, _Float16* __restrict__ VT)
{
  int tid = blockIdx.x * 256 + threadIdx.x;
  int d   = tid & 63;          // feature pair (d, d+64)
  int row = tid >> 6;          // b*T + t
  int b   = row >> 12;
  int t   = row & (TLEN - 1);
  int pc  = t >> 10;
  int l   = t & (LCHUNK - 1);

  float inv = powf(10000.f, -(float)d * (1.f / 64.f));
  float ang = (float)t * inv;
  float cs = cosf(ang), sn = sinf(ang);

  size_t qi = (size_t)t * (NHEAD * DDIM) + (size_t)b * DDIM + d;
  float q1 = q[qi], q2 = q[qi + 64];
  float k1 = k[qi], k2 = k[qi + 64];
  float v1 = v[qi], v2 = v[qi + 64];

  const float sc = 0.08838834764831845f;  // 1/sqrt(128)
  float qr1 = (q1 * cs - q2 * sn) * sc, qr2 = (q2 * cs + q1 * sn) * sc;
  float kr1 = (k1 * cs - k2 * sn),      kr2 = (k2 * cs + k1 * sn);

  size_t qo = ((size_t)b * TLEN + t) * DDIM + d;
  QH[qo] = (_Float16)qr1; QH[qo + 64] = (_Float16)qr2;

  size_t ko = (((size_t)b * PNUM + pc) * MLEN + EHIS + l) * DDIM + d;
  KL[ko] = (_Float16)kr1; KL[ko + 64] = (_Float16)kr2;

  size_t vo = (((size_t)b * PNUM + pc) * DDIM + d) * MLEN + EHIS + l;
  VT[vo] = (_Float16)v1; VT[vo + (size_t)64 * MLEN] = (_Float16)v2;

  if (l >= LCHUNK - EHIS && pc < PNUM - 1) {   // history copy into next chunk
    int l2 = l - (LCHUNK - EHIS);
    size_t ko2 = (((size_t)b * PNUM + pc + 1) * MLEN + l2) * DDIM + d;
    KL[ko2] = (_Float16)kr1; KL[ko2 + 64] = (_Float16)kr2;
    size_t vo2 = (((size_t)b * PNUM + pc + 1) * DDIM + d) * MLEN + l2;
    VT[vo2] = (_Float16)v1; VT[vo2 + (size_t)64 * MLEN] = (_Float16)v2;
  }
}

// ---------------- gkv: block-mean keys/values for global attention ----------------
__global__ __launch_bounds__(128) void gkv_kernel(
    const float* __restrict__ zk, const float* __restrict__ zv,
    const _Float16* __restrict__ KL, const _Float16* __restrict__ VT,
    _Float16* __restrict__ KG, _Float16* __restrict__ VGT)
{
  int b = blockIdx.x >> 5, kk = blockIdx.x & 31, d = threadIdx.x;
  if (kk == 0) {
    KG[(size_t)b * 32 * DDIM + d]       = (_Float16)zk[b * DDIM + d];
    VGT[((size_t)b * DDIM + d) * 32]    = (_Float16)zv[b * DDIM + d];
    return;
  }
  int sblk = kk - 1;
  int pc = sblk >> 3;
  int l0 = (sblk & 7) * 128;
  const _Float16* kr = KL + (((size_t)b * PNUM + pc) * MLEN + EHIS + l0) * DDIM + d;
  const _Float16* vr = VT + (((size_t)b * PNUM + pc) * DDIM + d) * MLEN + EHIS + l0;
  float ks = 0.f, vs = 0.f;
  for (int j = 0; j < 128; ++j) { ks += (float)kr[(size_t)j * DDIM]; vs += (float)vr[j]; }
  KG[((size_t)b * 32 + kk) * DDIM + d]    = (_Float16)(ks * (1.f / 128.f));
  VGT[((size_t)b * DDIM + d) * 32 + kk]   = (_Float16)(vs * (1.f / 128.f));
}

// ---------------- attn: flash local (16x1152, 64-key blocks) + fused global (16x32) ----------------
__global__ __launch_bounds__(128) void attn_kernel(
    const _Float16* __restrict__ QH, const _Float16* __restrict__ KL,
    const _Float16* __restrict__ VT, const _Float16* __restrict__ KG,
    const _Float16* __restrict__ VGT, float* __restrict__ out)
{
  __shared__ __align__(16) _Float16 plds[4][16][72];
  const int lane  = threadIdx.x & 31;
  const int wslot = threadIdx.x >> 5;
  const int wid   = (blockIdx.x << 2) + wslot;
  const int tile = wid & 63;
  const int pc   = (wid >> 6) & 3;
  const int b    = wid >> 8;
  const int l0 = tile << 4;
  const int t0 = (pc << 10) + l0;
  const int r  = lane & 15;           // A row / B col / C col
  const int bk = (lane >> 4) << 3;    // K-fragment base AND C row-group offset
  _Float16 (*pl)[72] = plds[wslot];

  // Q A-fragments (16x32 f16 layout: lane r holds K {bk..bk+7, bk+16..bk+23})
  const _Float16* qrow = QH + ((size_t)b * TLEN + t0 + r) * DDIM;
  v16h qA[4];
#pragma unroll
  for (int g = 0; g < 4; ++g)
    qA[g] = cat8(ld8(qrow + g * 32 + bk), ld8(qrow + g * 32 + bk + 16));

  const v8f zero8 = {0.f, 0.f, 0.f, 0.f, 0.f, 0.f, 0.f, 0.f};
  float m[8], lsum[8];
  v8f o[8];
#pragma unroll
  for (int v = 0; v < 8; ++v) { m[v] = -__builtin_inff(); lsum[v] = 0.f; }
#pragma unroll
  for (int f = 0; f < 8; ++f) o[f] = zero8;

  const _Float16* KLb = KL + ((size_t)b * PNUM + pc) * MLEN * DDIM;
  const _Float16* VTb = VT + ((size_t)b * PNUM + pc) * DDIM * MLEN;
  const int kb0   = (pc == 0) ? 2 : 0;               // chunk 0: history (keys 0..127) masked
  const int kbend = ((l0 + 143) >> 6) + 1;           // last visible key = l0+15+128

  for (int kb = kb0; kb < kbend; ++kb) {
    const int key0 = kb << 6;
    v8f s[4];
#pragma unroll
    for (int tt = 0; tt < 4; ++tt) s[tt] = zero8;
#pragma unroll
    for (int g = 0; g < 4; ++g) {                    // QK^T over D=128, 4 key sub-tiles
#pragma unroll
      for (int tt = 0; tt < 4; ++tt) {
        const _Float16* kp = KLb + (size_t)(key0 + tt * 16 + r) * DDIM + g * 32 + bk;
        s[tt] = wmma16(qA[g], cat8(ld8(kp), ld8(kp + 16)), s[tt]);
      }
    }
    if (key0 + 63 > l0 + 128) {                      // causal edge: key > row+E masked
#pragma unroll
      for (int tt = 0; tt < 4; ++tt) {
        const int col = key0 + tt * 16 + r;
#pragma unroll
        for (int v = 0; v < 8; ++v)
          if (col > l0 + v + bk + 128) s[tt][v] = -__builtin_inff();
      }
    }
    float sc8[8];
#pragma unroll
    for (int v = 0; v < 8; ++v) {                    // online softmax, rows = v+bk
      float mx = fmaxf(fmaxf(s[0][v], s[1][v]), fmaxf(s[2][v], s[3][v]));
      mx = rmax16(mx);
      float mn = fmaxf(m[v], mx);
      float scale = __expf(m[v] - mn);
      float p0 = __expf(s[0][v] - mn);
      float p1 = __expf(s[1][v] - mn);
      float p2 = __expf(s[2][v] - mn);
      float p3 = __expf(s[3][v] - mn);
      lsum[v] = lsum[v] * scale + rsum16((p0 + p1) + (p2 + p3));
      m[v] = mn;
      sc8[v] = scale;
      pl[v + bk][r]      = (_Float16)p0;             // C-layout -> LDS tile (16x64)
      pl[v + bk][16 + r] = (_Float16)p1;
      pl[v + bk][32 + r] = (_Float16)p2;
      pl[v + bk][48 + r] = (_Float16)p3;
    }
#pragma unroll
    for (int f = 0; f < 8; ++f)
#pragma unroll
      for (int v = 0; v < 8; ++v) o[f][v] *= sc8[v];
    __builtin_amdgcn_wave_barrier();                 // DS ops in-order within wave
    v16h pA0 = cat8(*(const v8h*)&pl[r][bk],      *(const v8h*)&pl[r][bk + 16]);
    v16h pA1 = cat8(*(const v8h*)&pl[r][bk + 32], *(const v8h*)&pl[r][bk + 48]);
    __builtin_amdgcn_wave_barrier();
#pragma unroll
    for (int f = 0; f < 8; ++f) {                    // P(16x64) x V(64x16) per feat tile
      const _Float16* vp = VTb + (size_t)(f * 16 + r) * MLEN + key0 + bk;
      o[f] = wmma16(pA0, cat8(ld8(vp),      ld8(vp + 16)), o[f]);
      o[f] = wmma16(pA1, cat8(ld8(vp + 32), ld8(vp + 48)), o[f]);
    }
    __builtin_amdgcn_wave_barrier();
  }

  // ---- global attention: 32 mean-keys, uniform visible count per 16-row tile ----
  const int nk = (t0 >> 7) + 1;                      // keys 0..nk-1 visible, nk<=32
  v8f g0 = zero8, g1 = zero8;
  const _Float16* KGb = KG + (size_t)b * 32 * DDIM;
#pragma unroll
  for (int g = 0; g < 4; ++g) {
    const _Float16* k0p = KGb + (size_t)r * DDIM + g * 32 + bk;
    const _Float16* k1p = KGb + (size_t)(16 + r) * DDIM + g * 32 + bk;
    g0 = wmma16(qA[g], cat8(ld8(k0p), ld8(k0p + 16)), g0);
    g1 = wmma16(qA[g], cat8(ld8(k1p), ld8(k1p + 16)), g1);
  }
  float gl[8];
#pragma unroll
  for (int v = 0; v < 8; ++v) {
    if (r >= nk)      g0[v] = -__builtin_inff();
    if (16 + r >= nk) g1[v] = -__builtin_inff();
    float mx = rmax16(fmaxf(g0[v], g1[v]));
    float p0 = __expf(g0[v] - mx);
    float p1 = __expf(g1[v] - mx);
    gl[v] = rsum16(p0 + p1);
    pl[v + bk][r]      = (_Float16)p0;
    pl[v + bk][16 + r] = (_Float16)p1;
  }
  __builtin_amdgcn_wave_barrier();
  v16h pgA = cat8(*(const v8h*)&pl[r][bk], *(const v8h*)&pl[r][bk + 16]);
  __builtin_amdgcn_wave_barrier();

  float il[8], ig[8];
#pragma unroll
  for (int v = 0; v < 8; ++v) { il[v] = 1.f / lsum[v]; ig[v] = 1.f / gl[v]; }

#pragma unroll
  for (int f = 0; f < 8; ++f) {
    const _Float16* vp = VGT + ((size_t)b * DDIM + f * 16 + r) * 32 + bk;
    v16h vB = cat8(ld8(vp), ld8(vp + 16));
    v8f og = wmma16(pgA, vB, zero8);
#pragma unroll
    for (int v = 0; v < 8; ++v) {
      const int row = v + bk;
      out[(size_t)(t0 + row) * (NHEAD * DDIM) + (size_t)b * DDIM + f * 16 + r] =
          o[f][v] * il[v] + og[v] * ig[v];
    }
  }
}

extern "C" void kernel_launch(void* const* d_in, const int* in_sizes, int n_in,
                              void* d_out, int out_size, void* d_ws, size_t ws_size,
                              hipStream_t stream)
{
  (void)in_sizes; (void)n_in; (void)out_size; (void)ws_size;
  const float* q  = (const float*)d_in[0];
  const float* k  = (const float*)d_in[1];
  const float* v  = (const float*)d_in[2];
  const float* zk = (const float*)d_in[3];
  const float* zv = (const float*)d_in[4];

  _Float16* QH  = (_Float16*)d_ws;                                   // 16*4096*128
  _Float16* KL  = QH + (size_t)NHEAD * TLEN * DDIM;                  // 16*4*1152*128
  _Float16* VT  = KL + (size_t)NHEAD * PNUM * MLEN * DDIM;           // 16*4*128*1152
  _Float16* KG  = VT + (size_t)NHEAD * PNUM * DDIM * MLEN;           // 16*32*128
  _Float16* VGT = KG + (size_t)NHEAD * 32 * DDIM;                    // 16*128*32

  prep_kernel<<<(NHEAD * TLEN * 64) / 256, 256, 0, stream>>>(q, k, v, QH, KL, VT);
  gkv_kernel<<<NHEAD * 32, 128, 0, stream>>>(zk, zv, KL, VT, KG, VGT);
  attn_kernel<<<(NHEAD * PNUM * (LCHUNK / 16)) / 4, 128, 0, stream>>>(QH, KL, VT, KG, VGT, (float*)d_out);
}
